// GRACELoss_68521908240439
// MI455X (gfx1250) — matrix operand order repaired and checked by reading, not compiled, not run on hardware
//
#include <hip/hip_runtime.h>
#include <hip/hip_bf16.h>

typedef _Float16 v8h  __attribute__((ext_vector_type(8)));
typedef _Float16 v16h __attribute__((ext_vector_type(16)));
typedef float    v8f  __attribute__((ext_vector_type(8)));

#define DDIM  128
#define INV_T 2.0f   // 1 / TEMPERATURE, TEMPERATURE = 0.5

static __device__ __forceinline__ v16h cat8(v8h a, v8h b) {
    return __builtin_shufflevector(a, b, 0,1,2,3,4,5,6,7,8,9,10,11,12,13,14,15);
}

// ---------------------------------------------------------------------------
// Kernel 1: row-normalize h1,h2; emit f16 hi/lo splits of the normalized rows
// (x ~= hi + lo, |lo| <= 2^-11 |x|) plus exact f32 diagonal terms.
// One block (128 threads) per row; D == 128.
// ---------------------------------------------------------------------------
__global__ __launch_bounds__(128) void grace_norm_kernel(
    const float* __restrict__ h1, const float* __restrict__ h2,
    _Float16* __restrict__ ahi, _Float16* __restrict__ alo,
    _Float16* __restrict__ bhi, _Float16* __restrict__ blo,
    float* __restrict__ expAA, float* __restrict__ expBB,
    float* __restrict__ sAB)
{
    const int i    = blockIdx.x;
    const int t    = threadIdx.x;
    const int wave = t >> 5;
    const int lane = t & 31;

    const float x = h1[i * DDIM + t];
    const float y = h2[i * DDIM + t];

    __shared__ float sm[4];

    // sum(x*x)
    float v = x * x;
    #pragma unroll
    for (int m = 1; m < 32; m <<= 1) v += __shfl_xor(v, m, 32);
    if (lane == 0) sm[wave] = v;
    __syncthreads();
    const float s1 = sm[0] + sm[1] + sm[2] + sm[3];
    __syncthreads();

    // sum(y*y)
    v = y * y;
    #pragma unroll
    for (int m = 1; m < 32; m <<= 1) v += __shfl_xor(v, m, 32);
    if (lane == 0) sm[wave] = v;
    __syncthreads();
    const float s2 = sm[0] + sm[1] + sm[2] + sm[3];
    __syncthreads();

    const float inv1 = 1.0f / fmaxf(sqrtf(s1), 1e-8f);
    const float inv2 = 1.0f / fmaxf(sqrtf(s2), 1e-8f);
    const float av = x * inv1;
    const float bv = y * inv2;

    // hi/lo f16 split (hi + lo reproduces av to ~2^-22 relative)
    const _Float16 avh = (_Float16)av;
    const _Float16 avl = (_Float16)(av - (float)avh);
    const _Float16 bvh = (_Float16)bv;
    const _Float16 bvl = (_Float16)(bv - (float)bvh);
    ahi[i * DDIM + t] = avh;
    alo[i * DDIM + t] = avl;
    bhi[i * DDIM + t] = bvh;
    blo[i * DDIM + t] = bvl;

    // sum(a*b) for the between-sim diagonal (exact f32)
    v = av * bv;
    #pragma unroll
    for (int m = 1; m < 32; m <<= 1) v += __shfl_xor(v, m, 32);
    if (lane == 0) sm[wave] = v;
    __syncthreads();
    const float sab = sm[0] + sm[1] + sm[2] + sm[3];

    if (t == 0) {
        expAA[i] = __expf(s1 * inv1 * inv1 * INV_T);  // exp(a_i . a_i / T)
        expBB[i] = __expf(s2 * inv2 * inv2 * INV_T);  // exp(b_i . b_i / T)
        sAB[i]   = sab * INV_T;                       // (a_i . b_i) / T (raw)
    }
}

// ---------------------------------------------------------------------------
// Kernel 2: fused  rowsum_i( exp( (X @ Y^T)[i,:] / T ) )  via f16 hi/lo WMMA.
// grid.x = N/16 row tiles, grid.y = 4 selects (X,Y,out) pair.
// Block = 128 threads = 4 waves; each wave strides over column tiles.
//
// Per 16x16x128 tile: all 16 B loads are staged into registers first (one
// clause, partial loadcnt waits), then 4 K-chunks x 3 products
// (hiA*hiB, hiA*loB, loA*hiB) into three independent accumulators.
// Next tile's rows are prefetched (global_prefetch_b8) during compute.
//
// 16-bit WMMA operand layout (wave32, h = lane>>4):
//   A 16x32: lane holds K = {32c+8h .. +7} in elems 0-7, {32c+16+8h ..} in 8-15
//   B 32x16: lane holds K = {32c+8h .. +7} in elems 0-7, {32c+16+8h ..} in 8-15
//   C/D    : lane L holds column N=L%16, rows r+8*(L>>4) in VGPR r
// ---------------------------------------------------------------------------
__global__ __launch_bounds__(128) void grace_gemm_kernel(
    const _Float16* __restrict__ ahi, const _Float16* __restrict__ alo,
    const _Float16* __restrict__ bhi, const _Float16* __restrict__ blo,
    float* __restrict__ rsumA, float* __restrict__ rsumB,
    float* __restrict__ rsumAB, float* __restrict__ rsumBA, int n)
{
    const _Float16 *Xh, *Xl, *Yh, *Yl; float* out;
    switch (blockIdx.y) {
        case 0:  Xh = ahi; Xl = alo; Yh = ahi; Yl = alo; out = rsumA;  break;
        case 1:  Xh = bhi; Xl = blo; Yh = bhi; Yl = blo; out = rsumB;  break;
        case 2:  Xh = ahi; Xl = alo; Yh = bhi; Yl = blo; out = rsumAB; break;
        default: Xh = bhi; Xl = blo; Yh = ahi; Yl = alo; out = rsumBA; break;
    }

    const int tid  = threadIdx.x;
    const int wave = tid >> 5;
    const int lane = tid & 31;
    const int nt   = n >> 4;
    const int rowX = (blockIdx.x << 4) + (lane & 15);
    const int h    = lane >> 4;

    // Cache the 16x128 A stripe (hi and lo) in registers: 2 x 4 x v16h = 64 VGPRs.
    v16h Ah[4], Al[4];
    {
        const _Float16* xh = Xh + rowX * DDIM;
        const _Float16* xl = Xl + rowX * DDIM;
        #pragma unroll
        for (int c = 0; c < 4; ++c) {
            const int off = 32 * c + 8 * h;
            Ah[c] = cat8(*(const v8h*)(xh + off), *(const v8h*)(xh + off + 16));
            Al[c] = cat8(*(const v8h*)(xl + off), *(const v8h*)(xl + off + 16));
        }
    }

    v8f racc = {0.f, 0.f, 0.f, 0.f, 0.f, 0.f, 0.f, 0.f};

    for (int J = wave; J < nt; J += 4) {
        const int rowY = (J << 4) + (lane & 15);
        const _Float16* yh = Yh + rowY * DDIM + 8 * h;
        const _Float16* yl = Yl + rowY * DDIM + 8 * h;

        // Prefetch next tile: lanes l and l+16 share a row, so the lane half
        // selects which 128B line of the 256B row to pull.
        if (J + 4 < nt) {
            const int rowN = ((J + 4) << 4) + (lane & 15);
            __builtin_prefetch(Yh + rowN * DDIM + 64 * h, 0, 3);
            __builtin_prefetch(Yl + rowN * DDIM + 64 * h, 0, 3);
        }

        // Stage the full B tile (hi+lo) in registers: one load clause,
        // partial waits let later loads overlap early WMMAs.
        v16h Bh[4], Bl[4];
        #pragma unroll
        for (int c = 0; c < 4; ++c) {
            const int off = 32 * c;
            Bh[c] = cat8(*(const v8h*)(yh + off), *(const v8h*)(yh + off + 16));
            Bl[c] = cat8(*(const v8h*)(yl + off), *(const v8h*)(yl + off + 16));
        }

        v8f acc0 = {0.f, 0.f, 0.f, 0.f, 0.f, 0.f, 0.f, 0.f};
        v8f acc1 = {0.f, 0.f, 0.f, 0.f, 0.f, 0.f, 0.f, 0.f};
        v8f acc2 = {0.f, 0.f, 0.f, 0.f, 0.f, 0.f, 0.f, 0.f};
        #pragma unroll
        for (int c = 0; c < 4; ++c) {
            acc0 = __builtin_amdgcn_wmma_f32_16x16x32_f16(
                false, Ah[c], false, Bh[c], (short)0, acc0, false, false);
            acc1 = __builtin_amdgcn_wmma_f32_16x16x32_f16(
                false, Ah[c], false, Bl[c], (short)0, acc1, false, false);
            acc2 = __builtin_amdgcn_wmma_f32_16x16x32_f16(
                false, Al[c], false, Bh[c], (short)0, acc2, false, false);
        }
        #pragma unroll
        for (int r = 0; r < 8; ++r)
            racc[r] += __expf((acc0[r] + acc1[r] + acc2[r]) * INV_T);
    }

    // Row reduction: masks 1,2,4,8 keep lane halves separate
    // (lower half = rows 0-7, upper half = rows 8-15 of the tile).
    #pragma unroll
    for (int m = 1; m < 16; m <<= 1) {
        #pragma unroll
        for (int r = 0; r < 8; ++r)
            racc[r] += __shfl_xor(racc[r], m, 32);
    }

    __shared__ float sm[64];   // [wave][16 rows]
    if (lane == 0) {
        #pragma unroll
        for (int r = 0; r < 8; ++r) sm[wave * 16 + r] = racc[r];
    }
    if (lane == 16) {
        #pragma unroll
        for (int r = 0; r < 8; ++r) sm[wave * 16 + 8 + r] = racc[r];
    }
    __syncthreads();
    if (tid < 16) {
        const float s = sm[tid] + sm[16 + tid] + sm[32 + tid] + sm[48 + tid];
        out[(blockIdx.x << 4) + tid] = s;   // exclusive writer: no atomics
    }
}

// ---------------------------------------------------------------------------
// Kernel 3: per-row loss, mean, emit (loss, 1).
//   l1_i = log(rsumA + rsumAB - expAA) - sAB
//   l2_i = log(rsumB + rsumBA - expBB) - sAB
//   loss = mean(0.5*(l1+l2))
// ---------------------------------------------------------------------------
__global__ __launch_bounds__(256) void grace_final_kernel(
    const float* __restrict__ rsumA,  const float* __restrict__ rsumB,
    const float* __restrict__ rsumAB, const float* __restrict__ rsumBA,
    const float* __restrict__ expAA,  const float* __restrict__ expBB,
    const float* __restrict__ sAB, float* __restrict__ out, int n)
{
    const int wave = threadIdx.x >> 5;
    const int lane = threadIdx.x & 31;

    float p = 0.f;
    for (int i = threadIdx.x; i < n; i += 256) {
        const float d1 = rsumA[i] + rsumAB[i] - expAA[i];
        const float d2 = rsumB[i] + rsumBA[i] - expBB[i];
        p += 0.5f * (logf(d1) + logf(d2)) - sAB[i];
    }
    #pragma unroll
    for (int m = 1; m < 32; m <<= 1) p += __shfl_xor(p, m, 32);

    __shared__ float sm[8];
    if (lane == 0) sm[wave] = p;
    __syncthreads();
    if (threadIdx.x == 0) {
        float total = 0.f;
        #pragma unroll
        for (int w = 0; w < 8; ++w) total += sm[w];
        out[0] = total / (float)n;
        out[1] = 1.0f;
    }
}

// ---------------------------------------------------------------------------
extern "C" void kernel_launch(void* const* d_in, const int* in_sizes, int n_in,
                              void* d_out, int out_size, void* d_ws, size_t ws_size,
                              hipStream_t stream) {
    const float* h1 = (const float*)d_in[0];
    const float* h2 = (const float*)d_in[1];
    const int n = in_sizes[0] / DDIM;   // 8192

    _Float16* ahi = (_Float16*)d_ws;                 // n*128 f16
    _Float16* alo = ahi + (size_t)n * DDIM;          // n*128 f16
    _Float16* bhi = alo + (size_t)n * DDIM;          // n*128 f16
    _Float16* blo = bhi + (size_t)n * DDIM;          // n*128 f16
    float* expAA  = (float*)(blo + (size_t)n * DDIM);// n
    float* expBB  = expAA + n;                       // n
    float* sAB    = expBB + n;                       // n
    float* rsumA  = sAB + n;                         // n
    float* rsumB  = rsumA + n;                       // n
    float* rsumAB = rsumB + n;                       // n
    float* rsumBA = rsumAB + n;                      // n

    float* out = (float*)d_out;

    grace_norm_kernel<<<n, DDIM, 0, stream>>>(h1, h2, ahi, alo, bhi, blo,
                                              expAA, expBB, sAB);

    dim3 grid(n >> 4, 4);
    grace_gemm_kernel<<<grid, 128, 0, stream>>>(ahi, alo, bhi, blo,
                                                rsumA, rsumB, rsumAB, rsumBA, n);

    grace_final_kernel<<<1, 256, 0, stream>>>(rsumA, rsumB, rsumAB, rsumBA,
                                              expAA, expBB, sAB, out, n);
}
